// MambaBlock_80187039416644
// MI455X (gfx1250) — matrix operand order repaired and checked by reading, compiled, not run on hardware
//
#include <hip/hip_runtime.h>
#include <stdint.h>

#define BB 8192
#define DD 512
#define SS 16

typedef __attribute__((ext_vector_type(16))) __bf16    v16bf;
typedef __attribute__((ext_vector_type(8)))  float     v8f;
typedef __attribute__((ext_vector_type(4)))  uint32_t  u32x4;
typedef __attribute__((ext_vector_type(4)))  float     f32x4;

struct Frag { u32x4 lo, hi; };   // 32 bytes == v16bf

__device__ __forceinline__ v16bf frag_cast(Frag f) {
    return __builtin_bit_cast(v16bf, f);
}

__device__ __forceinline__ uint16_t f32_to_bf16(float f) {
    uint32_t u = __float_as_uint(f);
    uint32_t r = u + 0x7FFFu + ((u >> 16) & 1u);   // round-to-nearest-even
    return (uint16_t)(r >> 16);
}

__device__ __forceinline__ float silu_f(float x)     { return x / (1.0f + __expf(-x)); }
__device__ __forceinline__ float softplus_f(float x) { return (x > 20.0f) ? x : log1pf(__expf(x)); }

// ---------------------------------------------------------------- converts
__global__ void k_f32_to_bf16(const float* __restrict__ in, uint16_t* __restrict__ out, int n) {
    int i = blockIdx.x * blockDim.x + threadIdx.x;
    if (i < n) out[i] = f32_to_bf16(in[i]);
}

__global__ void k_prep_A(const float* __restrict__ A_log, float* __restrict__ A, int n) {
    int i = blockIdx.x * blockDim.x + threadIdx.x;
    if (i < n) A[i] = -__expf(A_log[i]);
}

// ---------------------------------------------------------------- layernorm -> bf16
__global__ __launch_bounds__(256) void k_layernorm(const float* __restrict__ x,
                                                   const float* __restrict__ g,
                                                   const float* __restrict__ b,
                                                   uint16_t* __restrict__ xn) {
    __shared__ float s_sum[8], s_sq[8];
    const int row = blockIdx.x;
    const int t   = threadIdx.x;                 // 256 threads, 2 elems each
    const float* xr = x + (size_t)row * DD;
    float2 v2 = *reinterpret_cast<const float2*>(xr + t * 2);
    float s  = v2.x + v2.y;
    float sq = v2.x * v2.x + v2.y * v2.y;
#pragma unroll
    for (int o = 16; o > 0; o >>= 1) {
        s  += __shfl_down(s,  o, 32);
        sq += __shfl_down(sq, o, 32);
    }
    const int lane = t & 31, w = t >> 5;
    if (lane == 0) { s_sum[w] = s; s_sq[w] = sq; }
    __syncthreads();
    float ts = 0.f, tq = 0.f;
#pragma unroll
    for (int i = 0; i < 8; ++i) { ts += s_sum[i]; tq += s_sq[i]; }
    const float mu   = ts * (1.0f / DD);
    const float var  = tq * (1.0f / DD) - mu * mu;
    const float rstd = rsqrtf(var + 1e-5f);
    const int d0 = t * 2;
    float ox = (v2.x - mu) * rstd * g[d0]     + b[d0];
    float oy = (v2.y - mu) * rstd * g[d0 + 1] + b[d0 + 1];
    uint32_t packed = (uint32_t)f32_to_bf16(ox) | ((uint32_t)f32_to_bf16(oy) << 16);
    *reinterpret_cast<uint32_t*>(xn + (size_t)row * DD + d0) = packed;
}

// ---------------------------------------------------------------- WMMA GEMM  out = A[M,K] @ W[N,K]^T
// Double-buffered fragment pipeline: prefetch k+32 before issuing current WMMAs.
// EPI: 0 = silu((acc+bias)*conv_center)  (writes f32 + bf16)
//      1 = silu(acc+bias)
//      2 = softplus(acc+bias)
//      3 = acc+bias
template <int NT, int EPI>
__global__ __launch_bounds__(256)
void k_wmma_gemm(const uint16_t* __restrict__ A, const uint16_t* __restrict__ W,
                 const float* __restrict__ bias, const float* __restrict__ conv3,
                 float* __restrict__ outf, uint16_t* __restrict__ outh,
                 int M, int N, int K) {
    const int lane   = threadIdx.x & 31;
    const int wave   = threadIdx.x >> 5;
    const int tile_m = blockIdx.y * 128 + wave * 16;
    const int tile_n = blockIdx.x * (16 * NT);
    const int n0     = lane & 15;
    const int ar  = tile_m + n0;
    const int akb = (lane >> 4) << 3;   // A: k-half base 0 / 8 (pairs 0-7&16-23 vs 8-15&24-31)
    const int bkb = (lane >> 4) << 3;   // B: k-half base 0 / 8 (pairs 0-7&16-23 vs 8-15&24-31)

    v8f acc[NT] = {};
    const uint16_t* ap = A + (size_t)ar * K + akb;
    const uint16_t* bp[NT];
#pragma unroll
    for (int t = 0; t < NT; ++t)
        bp[t] = W + (size_t)(tile_n + t * 16 + n0) * K + bkb;

    // prologue: fragments for k = 0
    Frag a_c, b_c[NT];
    a_c.lo = *reinterpret_cast<const u32x4*>(ap);
    a_c.hi = *reinterpret_cast<const u32x4*>(ap + 16);
#pragma unroll
    for (int t = 0; t < NT; ++t) {
        b_c[t].lo = *reinterpret_cast<const u32x4*>(bp[t]);
        b_c[t].hi = *reinterpret_cast<const u32x4*>(bp[t] + 16);
    }

#pragma unroll 2
    for (int k0 = 0; k0 < K; k0 += 32) {
        // prefetch next iteration (clamped: last iteration harmlessly reloads itself)
        const int k1 = (k0 + 32 < K) ? (k0 + 32) : k0;
        Frag a_n, b_n[NT];
        a_n.lo = *reinterpret_cast<const u32x4*>(ap + k1);
        a_n.hi = *reinterpret_cast<const u32x4*>(ap + k1 + 16);
#pragma unroll
        for (int t = 0; t < NT; ++t) {
            b_n[t].lo = *reinterpret_cast<const u32x4*>(bp[t] + k1);
            b_n[t].hi = *reinterpret_cast<const u32x4*>(bp[t] + k1 + 16);
        }
        // compute on current fragments (loads above are still in flight)
        v16bf av = frag_cast(a_c);
#pragma unroll
        for (int t = 0; t < NT; ++t)
            acc[t] = __builtin_amdgcn_wmma_f32_16x16x32_bf16(
                false, av, false, frag_cast(b_c[t]), (short)0, acc[t], false, false);
        a_c = a_n;
#pragma unroll
        for (int t = 0; t < NT; ++t) b_c[t] = b_n[t];
    }

    const int m_base = tile_m + ((lane >> 4) << 3);
#pragma unroll
    for (int t = 0; t < NT; ++t) {
        const int   n  = tile_n + t * 16 + n0;
        const float bn = bias[n];
        const float sc = (EPI == 0) ? conv3[n * 3 + 1] : 0.f;
#pragma unroll
        for (int v = 0; v < 8; ++v) {
            float val = acc[t][v] + bn;
            if      (EPI == 0) val = silu_f(val * sc);
            else if (EPI == 1) val = silu_f(val);
            else if (EPI == 2) val = softplus_f(val);
            size_t off = (size_t)(m_base + v) * N + n;
            if (outf) outf[off] = val;
            if (outh) outh[off] = f32_to_bf16(val);
        }
    }
}

// ---------------------------------------------------------------- SSM state update (roofline pass)
// 512 MB of state traffic is non-temporal (single use); the freshly produced
// delta/x_conv/v activations stay regular-temporal so they hit in the 192 MB L2.
__global__ __launch_bounds__(256)
void k_ssm_state(const float* __restrict__ st_in, const float* __restrict__ delta,
                 const float* __restrict__ xc,    const float* __restrict__ Bm,
                 const float* __restrict__ Cm,    const float* __restrict__ A,
                 const float* __restrict__ vgate, float* __restrict__ st_out,
                 uint16_t* __restrict__ ybf) {
    const int idx = blockIdx.x * blockDim.x + threadIdx.x;   // over B*D (exact)
    const int b = idx >> 9;            // / DD
    const int d = idx & (DD - 1);      // % DD
    const float dl = delta[idx];
    const float dx = dl * xc[idx];
    const f32x4* sp = reinterpret_cast<const f32x4*>(st_in + (size_t)idx * SS);
    const f32x4* al = reinterpret_cast<const f32x4*>(A     + (size_t)d   * SS);
    const f32x4* bp = reinterpret_cast<const f32x4*>(Bm    + (size_t)b   * SS);
    const f32x4* cp = reinterpret_cast<const f32x4*>(Cm    + (size_t)b   * SS);
    f32x4*       op = reinterpret_cast<f32x4*>(st_out + (size_t)idx * SS);
    float accum = 0.f;
#pragma unroll
    for (int q = 0; q < 4; ++q) {
        f32x4 s4 = __builtin_nontemporal_load(sp + q);
        f32x4 a4 = al[q], b4 = bp[q], c4 = cp[q];
        f32x4 ns;
        ns.x = s4.x * __expf(dl * a4.x) + dx * b4.x;
        ns.y = s4.y * __expf(dl * a4.y) + dx * b4.y;
        ns.z = s4.z * __expf(dl * a4.z) + dx * b4.z;
        ns.w = s4.w * __expf(dl * a4.w) + dx * b4.w;
        accum += ns.x * c4.x + ns.y * c4.y + ns.z * c4.z + ns.w * c4.w;
        __builtin_nontemporal_store(ns, op + q);
    }
    ybf[idx] = f32_to_bf16(accum * vgate[idx]);
}

// ---------------------------------------------------------------- launch
extern "C" void kernel_launch(void* const* d_in, const int* in_sizes, int n_in,
                              void* d_out, int out_size, void* d_ws, size_t ws_size,
                              hipStream_t stream) {
    const float* x     = (const float*)d_in[0];
    const float* st    = (const float*)d_in[1];
    const float* ln_g  = (const float*)d_in[2];
    const float* ln_b  = (const float*)d_in[3];
    const float* w1W   = (const float*)d_in[4];
    const float* w1b   = (const float*)d_in[5];
    const float* v1W   = (const float*)d_in[6];
    const float* v1b   = (const float*)d_in[7];
    const float* w2W   = (const float*)d_in[8];
    const float* w2b   = (const float*)d_in[9];
    const float* convw = (const float*)d_in[10];
    const float* A_log = (const float*)d_in[11];
    const float* BpW   = (const float*)d_in[12];
    const float* Bpb   = (const float*)d_in[13];
    const float* CpW   = (const float*)d_in[14];
    const float* Cpb   = (const float*)d_in[15];
    const float* dtW   = (const float*)d_in[16];
    const float* dtb   = (const float*)d_in[17];
    (void)in_sizes; (void)n_in; (void)out_size; (void)ws_size;

    float* out    = (float*)d_out;                 // [B,D]
    float* nstate = out + (size_t)BB * DD;         // [B,D,S]

    char*  ws  = (char*)d_ws;
    size_t off = 0;
    auto alloc = [&](size_t bytes) -> char* {
        off = (off + 255) & ~(size_t)255;
        char* p = ws + off;
        off += bytes;
        return p;
    };
    uint16_t* w1h = (uint16_t*)alloc((size_t)DD * DD * 2);
    uint16_t* v1h = (uint16_t*)alloc((size_t)DD * DD * 2);
    uint16_t* dth = (uint16_t*)alloc((size_t)DD * DD * 2);
    uint16_t* w2h = (uint16_t*)alloc((size_t)DD * DD * 2);
    uint16_t* Bph = (uint16_t*)alloc((size_t)SS * DD * 2);
    uint16_t* Cph = (uint16_t*)alloc((size_t)SS * DD * 2);
    float*    Af  = (float*)   alloc((size_t)DD * SS * 4);
    uint16_t* xnh = (uint16_t*)alloc((size_t)BB * DD * 2);
    float*    xcf = (float*)   alloc((size_t)BB * DD * 4);
    uint16_t* xch = (uint16_t*)alloc((size_t)BB * DD * 2);
    float*    vf  = (float*)   alloc((size_t)BB * DD * 4);
    float*    dlf = (float*)   alloc((size_t)BB * DD * 4);
    float*    Bmf = (float*)   alloc((size_t)BB * SS * 4);
    float*    Cmf = (float*)   alloc((size_t)BB * SS * 4);
    uint16_t* yh  = (uint16_t*)alloc((size_t)BB * DD * 2);

    const int NW = DD * DD;   // 262144
    const int NS = SS * DD;   // 8192
    k_f32_to_bf16<<<(NW + 255) / 256, 256, 0, stream>>>(w1W, w1h, NW);
    k_f32_to_bf16<<<(NW + 255) / 256, 256, 0, stream>>>(v1W, v1h, NW);
    k_f32_to_bf16<<<(NW + 255) / 256, 256, 0, stream>>>(dtW, dth, NW);
    k_f32_to_bf16<<<(NW + 255) / 256, 256, 0, stream>>>(w2W, w2h, NW);
    k_f32_to_bf16<<<(NS + 255) / 256, 256, 0, stream>>>(BpW, Bph, NS);
    k_f32_to_bf16<<<(NS + 255) / 256, 256, 0, stream>>>(CpW, Cph, NS);
    k_prep_A<<<(DD * SS + 255) / 256, 256, 0, stream>>>(A_log, Af, DD * SS);

    k_layernorm<<<BB, 256, 0, stream>>>(x, ln_g, ln_b, xnh);

    dim3 gBig(DD / 64, BB / 128);   // (8, 64): 16x64 per wave, 128x64 per block
    dim3 gSm(1, BB / 128);          // N = 16
    k_wmma_gemm<4, 0><<<gBig, 256, 0, stream>>>(xnh, w1h, w1b, convw, xcf, xch, BB, DD, DD);
    k_wmma_gemm<4, 1><<<gBig, 256, 0, stream>>>(xnh, v1h, v1b, nullptr, vf, nullptr, BB, DD, DD);
    k_wmma_gemm<4, 2><<<gBig, 256, 0, stream>>>(xch, dth, dtb, nullptr, dlf, nullptr, BB, DD, DD);
    k_wmma_gemm<1, 3><<<gSm,  256, 0, stream>>>(xch, Bph, Bpb, nullptr, Bmf, nullptr, BB, SS, DD);
    k_wmma_gemm<1, 3><<<gSm,  256, 0, stream>>>(xch, Cph, Cpb, nullptr, Cmf, nullptr, BB, SS, DD);

    k_ssm_state<<<(BB * DD) / 256, 256, 0, stream>>>(st, dlf, xcf, Bmf, Cmf, Af, vf, nstate, yh);

    k_wmma_gemm<4, 3><<<gBig, 256, 0, stream>>>(yh, w2h, w2b, nullptr, out, nullptr, BB, DD, DD);
}
